// MultiHeadAttention_19215683682353
// MI455X (gfx1250) — matrix operand order repaired and checked by reading, not compiled, not run on hardware
//
#include <hip/hip_runtime.h>
#include <hip/hip_bf16.h>
#include <math.h>

typedef __attribute__((ext_vector_type(16))) _Float16 v16h;
typedef __attribute__((ext_vector_type(8)))  _Float16 v8h;
typedef __attribute__((ext_vector_type(8)))  float    v8f;
typedef __attribute__((ext_vector_type(4)))  unsigned int u32x4;
typedef __attribute__((ext_vector_type(8)))  int      i32x8;
typedef __attribute__((ext_vector_type(4)))  int      i32x4;

#define WMMA_F32_F16(a, b, c) \
  __builtin_amdgcn_wmma_f32_16x16x32_f16(false, (a), false, (b), (short)0, (c), false, false)

constexpr int Bc = 2;
constexpr int Sq = 2048;
constexpr int E  = 1024;
constexpr int H  = 16;
constexpr int Dh = 64;
constexpr int BS = Bc * Sq;          // 4096 tokens

__device__ __forceinline__ v8f vzero8() {
  v8f v;
#pragma unroll
  for (int i = 0; i < 8; ++i) v[i] = 0.0f;
  return v;
}

// A-fragment loader: 16x32 f16 tile, row-major source (row m, leading dim ld).
// lane l: m = l&15, hi = l>>4; elems 0..7 = K hi*8+0..7 ; elems 8..15 = K 16+hi*8+0..7
__device__ __forceinline__ v16h load_a_frag(const _Float16* base, int ld) {
  const int l = threadIdx.x & 31;
  const int m = l & 15, hi = l >> 4;
  const _Float16* p = base + (size_t)m * ld + hi * 8;
  v16h r;
  *(v8h*)&r       = *(const v8h*)p;
  *((v8h*)&r + 1) = *(const v8h*)(p + 16);
  return r;
}

// B-fragment loader: B[k][n] = T[n][k] with T row-major (16 rows n, 32 cols k, ld).
// lane l: n = l&15, h = l>>4; elems 0..7 = K h*8+0..7 ; elems 8..15 = K 16+h*8+0..7
__device__ __forceinline__ v16h load_b_fragT(const _Float16* base, int ld) {
  const int l = threadIdx.x & 31;
  const int n = l & 15, h = l >> 4;
  const _Float16* p = base + (size_t)n * ld + h * 8;
  v16h r;
  *(v8h*)&r       = *(const v8h*)p;
  *((v8h*)&r + 1) = *(const v8h*)(p + 16);
  return r;
}

// TDM 2-D tile load global->LDS (D# per CDNA5 ISA).
// Sizes/strides in elements (data_size = 2 bytes). Groups 2/3 zero (2-D tensor).
// This toolchain's builtin is the 6-arg form: (g0, g1, g2, g3, g4, cpol).
__device__ __forceinline__ void tdm_load_2d(unsigned int lds_addr,
                                            unsigned long long gaddr,
                                            unsigned int tensor_d0,
                                            unsigned int tensor_d1,
                                            unsigned int tile_d0,
                                            unsigned int tile_d1,
                                            unsigned long long stride0) {
  u32x4 g0;
  g0[0] = 1u;                                            // count=1, user D#
  g0[1] = lds_addr;                                      // LDS byte address
  g0[2] = (unsigned int)(gaddr & 0xffffffffu);           // global_addr[31:0]
  g0[3] = (unsigned int)((gaddr >> 32) & 0x1ffffffu)     // global_addr[56:32]
          | (2u << 30);                                  // type=2 ("image")
  i32x8 g1;
  g1[0] = (int)(1u << 16);                               // wg_mask=0, data_size=1 (2B)
  g1[1] = (int)((tensor_d0 & 0xffffu) << 16);            // tensor_dim0[15:0]
  g1[2] = (int)((tensor_d0 >> 16) | ((tensor_d1 & 0xffffu) << 16));
  g1[3] = (int)((tensor_d1 >> 16) | (tile_d0 << 16));    // tile_dim0
  g1[4] = (int)(tile_d1 & 0xffffu);                      // tile_dim1 (tile_dim2=0)
  g1[5] = (int)(stride0 & 0xffffffffu);                  // dim0 stride[31:0]
  g1[6] = (int)((stride0 >> 32) & 0xffffu);              // dim0 stride[47:32]
  g1[7] = 0;
  i32x4 gz4;
  gz4[0] = 0; gz4[1] = 0; gz4[2] = 0; gz4[3] = 0;
  i32x8 gz8;
#pragma unroll
  for (int i = 0; i < 8; ++i) gz8[i] = 0;
  __builtin_amdgcn_tensor_load_to_lds(g0, g1, gz4, gz4, gz8, 0);
}

// ---------------------------------------------------------------- kernel 1
// f32 -> f16 conversion of x and the three weight matrices.
__global__ void prep_f16(const float* __restrict__ x,
                         const float* __restrict__ Wq,
                         const float* __restrict__ Wk,
                         const float* __restrict__ Wv,
                         _Float16* __restrict__ xh,
                         _Float16* __restrict__ Wh) {
  const int NX = BS * E;        // 4,194,304
  const int NW = E * E;         // 1,048,576 (1<<20)
  int idx = blockIdx.x * 256 + threadIdx.x;
  if (idx < NX) {
    xh[idx] = (_Float16)x[idx];
  } else {
    int j = idx - NX;           // 0 .. 3*NW-1
    int w = j >> 20;
    int o = j & (NW - 1);
    const float* W = (w == 0) ? Wq : ((w == 1) ? Wk : Wv);
    Wh[j] = (_Float16)W[o];
  }
}

// ---------------------------------------------------------------- kernel 2
// QKV projection: out = x @ W^T + bias, written f16 TRANSPOSED as [b,h,d,s].
// One wave -> 16 rows x 64 cols (4 C-frags). K-loop unrolled x2 as a register
// ping-pong: each WMMA consumes its own fragment set, no copies, loads for the
// other set stay in flight under the matrix ops.
__global__ __launch_bounds__(256) void qkv_gemm(
    const _Float16* __restrict__ xh, const _Float16* __restrict__ Wh,
    const float* __restrict__ bq, const float* __restrict__ bk,
    const float* __restrict__ bv,
    _Float16* __restrict__ Qt, _Float16* __restrict__ Kt,
    _Float16* __restrict__ Vt) {
  const int wv   = threadIdx.x >> 5;
  const int lane = threadIdx.x & 31;
  const int m = lane & 15, hi = lane >> 4;

  const int gw = blockIdx.x * 8 + wv;     // 0 .. 12287
  const int cg = gw & 15;                 // 64-col group (E/64 = 16)
  const int rt = (gw >> 4) & 255;         // 16-row tile (BS/16 = 256)
  const int p  = gw >> 12;                // 0=Q 1=K 2=V

  const _Float16* W = Wh + (size_t)p * E * E;
  const _Float16* Abase = xh + (size_t)(rt * 16) * E;
  const _Float16* Bbase = W + (size_t)(cg * 64) * E;

  v8f acc[4];
#pragma unroll
  for (int j = 0; j < 4; ++j) acc[j] = vzero8();

  // stage 0 preload
  v16h a0 = load_a_frag(Abase, E);
  v16h b0[4];
#pragma unroll
  for (int j = 0; j < 4; ++j)
    b0[j] = load_b_fragT(Bbase + (size_t)(j * 16) * E, E);

  for (int kk = 0; kk < E; kk += 64) {
    // preload stage 1 (kk+32 always < E inside the loop)
    v16h a1 = load_a_frag(Abase + kk + 32, E);
    v16h b1[4];
#pragma unroll
    for (int j = 0; j < 4; ++j)
      b1[j] = load_b_fragT(Bbase + (size_t)(j * 16) * E + kk + 32, E);

#pragma unroll
    for (int j = 0; j < 4; ++j)
      acc[j] = WMMA_F32_F16(a0, b0[j], acc[j]);

    if (kk + 64 < E) {                     // preload next stage 0 (uniform)
      a0 = load_a_frag(Abase + kk + 64, E);
#pragma unroll
      for (int j = 0; j < 4; ++j)
        b0[j] = load_b_fragT(Bbase + (size_t)(j * 16) * E + kk + 64, E);
    }

#pragma unroll
    for (int j = 0; j < 4; ++j)
      acc[j] = WMMA_F32_F16(a1, b1[j], acc[j]);
  }

  const float* bias = (p == 0) ? bq : ((p == 1) ? bk : bv);
  _Float16*    Out  = (p == 0) ? Qt : ((p == 1) ? Kt : Vt);
  const int n0 = rt * 16;                 // token base
  const int bb = n0 >> 11;                // / Sq
  const int s0 = (n0 & (Sq - 1)) + 8 * hi;
#pragma unroll
  for (int j = 0; j < 4; ++j) {
    const int i  = cg * 64 + j * 16 + m;  // output feature (lane column)
    const float bi = bias[i];
    const int h = i >> 6, d = i & 63;
    v8h ph;
#pragma unroll
    for (int r = 0; r < 8; ++r) ph[r] = (_Float16)(acc[j][r] + bi);
    // transposed store: slot r -> consecutive s -> one packed 16B store
    *(v8h*)(Out + ((size_t)(bb * H + h) * Dh + d) * Sq + s0) = ph;
  }
}

// ---------------------------------------------------------------- kernel 3
// RoPE on Q and K; reads [b,h,d,s], writes [b,h,s,d] (the A/B-frag layout).
__global__ void rope_kernel(const _Float16* __restrict__ Qt,
                            const _Float16* __restrict__ Kt,
                            _Float16* __restrict__ Qr,
                            _Float16* __restrict__ Kr) {
  const int N1 = Bc * H * Dh * Sq;        // 4,194,304
  int idx = blockIdx.x * 256 + threadIdx.x;
  const bool isQ = idx < N1;
  const _Float16* In  = isQ ? Qt : Kt;
  _Float16*       Out = isQ ? Qr : Kr;
  int t = isQ ? idx : idx - N1;
  const int s  = t & (Sq - 1);
  const int d  = (t >> 11) & (Dh - 1);
  const int bh = t >> 17;
  const float v  = (float)In[((size_t)bh * Dh + d) * Sq + s];
  const int   dp = (d < 32) ? d + 32 : d - 32;
  const float vp = (float)In[((size_t)bh * Dh + dp) * Sq + s];
  const float sign = (d < 32) ? -1.0f : 1.0f;
  const int   j = d & 31;
  const float inv = powf(10000.0f, -(float)j * (1.0f / 32.0f));
  const float ang = (float)s * inv;
  float sn, cs;
  sincosf(ang, &sn, &cs);
  Out[((size_t)bh * Sq + s) * Dh + d] = (_Float16)(v * cs + sign * vp * sn);
}

// ---------------------------------------------------------------- kernel 4
// Flash attention. Block = 8 waves; each wave owns 16 query rows.
// K/V tiles staged by the Tensor Data Mover into DOUBLE-BUFFERED LDS:
// wave 0 issues tile t+1's two D# loads, then waits tensorcnt<=2 so tile t is
// ready while tile t+1's DMA overlaps the whole compute phase.
__global__ __launch_bounds__(256) void attn_kernel(
    const _Float16* __restrict__ Qr, const _Float16* __restrict__ Kr,
    const _Float16* __restrict__ Vt, const float* __restrict__ mask,
    float* __restrict__ out) {
  __shared__ __align__(16) _Float16 lK[2][32 * 64];  // [buf][key][d]
  __shared__ __align__(16) _Float16 lV[2][64 * 32];  // [buf][d][key]
  __shared__ __align__(16) _Float16 lP[8][16 * 32];  // per-wave P bounce

  const int tid = threadIdx.x, wv = tid >> 5, lane = tid & 31;
  const int m = lane & 15, hi = lane >> 4;
  const int nblkq = Sq / 128;             // 16
  const int bh = blockIdx.x / nblkq;
  const int qb = blockIdx.x % nblkq;
  const int b  = bh >> 4;
  const int hh = bh & 15;
  const int q0 = qb * 128 + wv * 16;

  const _Float16* Qbase = Qr + (size_t)bh * Sq * Dh;
  const _Float16* Kbase = Kr + (size_t)bh * Sq * Dh;
  const _Float16* Vbase = Vt + (size_t)bh * Dh * Sq;
  unsigned int lKa[2], lVa[2];
#pragma unroll
  for (int i = 0; i < 2; ++i) {
    lKa[i] = (unsigned int)(unsigned long long)(uintptr_t)&lK[i][0];
    lVa[i] = (unsigned int)(unsigned long long)(uintptr_t)&lV[i][0];
  }

  // Q fragments (16 rows x 64 d = two K=32 A-frags), kept in registers.
  v16h qf[2];
#pragma unroll
  for (int kk = 0; kk < 2; ++kk)
    qf[kk] = load_a_frag(Qbase + (size_t)q0 * Dh + kk * 32, Dh);

  v8f O[4];
#pragma unroll
  for (int j = 0; j < 4; ++j) O[j] = vzero8();
  v8f mrow, lrow;
#pragma unroll
  for (int r = 0; r < 8; ++r) { mrow[r] = -1e30f; lrow[r] = 0.0f; }
  const float scale = 0.125f;             // 1/sqrt(64)
  const int NT = Sq / 32;                 // 64 key tiles

  // prologue: stage tile 0 into buffer 0
  if (wv == 0) {
    tdm_load_2d(lKa[0], (unsigned long long)(uintptr_t)Kbase,
                Dh, Sq, Dh, 32, Dh);
    tdm_load_2d(lVa[0], (unsigned long long)(uintptr_t)Vbase,
                Sq, Dh, 32, Dh, Sq);
  }

  for (int t = 0; t < NT; ++t) {
    const int buf = t & 1;
    const int k0  = t * 32;
    if (wv == 0) {
      if (t + 1 < NT) {                   // issue next tile into other buffer
        const int kn = k0 + 32;
        tdm_load_2d(lKa[buf ^ 1],
                    (unsigned long long)(uintptr_t)(Kbase + (size_t)kn * Dh),
                    Dh, Sq, Dh, 32, Dh);
        tdm_load_2d(lVa[buf ^ 1],
                    (unsigned long long)(uintptr_t)(Vbase + kn),
                    Sq, Dh, 32, Dh, Sq);
        __builtin_amdgcn_s_wait_tensorcnt(2);  // current tile's 2 DMAs done
      } else {
        __builtin_amdgcn_s_wait_tensorcnt(0);
      }
    }
    __syncthreads();
    const _Float16* Ktile = &lK[buf][0];
    const _Float16* Vtile = &lV[buf][0];

    // scores: two 16x16 C-frags (keys k0..k0+15, k0+16..k0+31)
    v8f s0 = vzero8(), s1 = vzero8();
#pragma unroll
    for (int kk = 0; kk < 2; ++kk) {
      v16h b0 = load_b_fragT(Ktile + 0 * 64 + kk * 32, 64);
      s0 = WMMA_F32_F16(qf[kk], b0, s0);
      v16h b1 = load_b_fragT(Ktile + 16 * 64 + kk * 32, 64);
      s1 = WMMA_F32_F16(qf[kk], b1, s1);
    }

    const float mk0 = mask[b * Sq + k0 + m];
    const float mk1 = mask[b * Sq + k0 + 16 + m];
    v8f mnew, corr;
#pragma unroll
    for (int r = 0; r < 8; ++r) {
      s0[r] = s0[r] * scale + mk0;
      s1[r] = s1[r] * scale + mk1;
      float tv = fmaxf(s0[r], s1[r]);
#pragma unroll
      for (int off = 1; off < 16; off <<= 1)
        tv = fmaxf(tv, __shfl_xor(tv, off, 16));
      mnew[r] = fmaxf(mrow[r], tv);
      corr[r] = __expf(mrow[r] - mnew[r]);
    }
#pragma unroll
    for (int r = 0; r < 8; ++r) {
      s0[r] = __expf(s0[r] - mnew[r]);
      s1[r] = __expf(s1[r] - mnew[r]);
      float tv = s0[r] + s1[r];
#pragma unroll
      for (int off = 1; off < 16; off <<= 1)
        tv += __shfl_xor(tv, off, 16);
      lrow[r] = lrow[r] * corr[r] + tv;
      mrow[r] = mnew[r];
    }
#pragma unroll
    for (int j = 0; j < 4; ++j)
#pragma unroll
      for (int r = 0; r < 8; ++r) O[j][r] *= corr[r];

    // P: C-layout -> A-layout via per-wave LDS bounce (wave-internal order,
    // LDS ops from one wave are in-order; just wait for DScnt).
    _Float16* lp = lP[wv];
#pragma unroll
    for (int r = 0; r < 8; ++r) {
      const int row = r + 8 * hi;
      lp[row * 32 + m]      = (_Float16)s0[r];
      lp[row * 32 + 16 + m] = (_Float16)s1[r];
    }
    asm volatile("s_wait_dscnt 0x0" ::: "memory");
    v16h pf = load_a_frag(lp, 32);

    // O += P @ V  (V B-frags contiguous thanks to [d][s] layout)
#pragma unroll
    for (int j = 0; j < 4; ++j) {
      v16h bV = load_b_fragT(Vtile + (j * 16) * 32, 32);
      O[j] = WMMA_F32_F16(pf, bV, O[j]);
    }
    __syncthreads();
  }

  // normalize by row sums and store f32 output [b, s, h*64+d]
  v8f rinv;
#pragma unroll
  for (int r = 0; r < 8; ++r) rinv[r] = 1.0f / lrow[r];
#pragma unroll
  for (int j = 0; j < 4; ++j) {
    const int d = j * 16 + m;
#pragma unroll
    for (int r = 0; r < 8; ++r) {
      const int srow = q0 + r + 8 * hi;
      out[(size_t)(b * Sq + srow) * E + hh * Dh + d] = O[j][r] * rinv[r];
    }
  }
}

// ---------------------------------------------------------------- launcher
extern "C" void kernel_launch(void* const* d_in, const int* in_sizes, int n_in,
                              void* d_out, int out_size, void* d_ws, size_t ws_size,
                              hipStream_t stream) {
  const float* x    = (const float*)d_in[0];
  const float* amsk = (const float*)d_in[1];
  const float* Wq   = (const float*)d_in[2];
  const float* bq   = (const float*)d_in[3];
  const float* Wk   = (const float*)d_in[4];
  const float* bk   = (const float*)d_in[5];
  const float* Wv   = (const float*)d_in[6];
  const float* bv   = (const float*)d_in[7];
  float* out = (float*)d_out;

  char* ws = (char*)d_ws;
  _Float16* xh = (_Float16*)ws; ws += (size_t)BS * E * 2;          // 8 MB
  _Float16* Wh = (_Float16*)ws; ws += (size_t)3 * E * E * 2;       // 6 MB
  _Float16* Qt = (_Float16*)ws; ws += (size_t)BS * E * 2;          // 8 MB
  _Float16* Kt = (_Float16*)ws; ws += (size_t)BS * E * 2;          // 8 MB
  _Float16* Vt = (_Float16*)ws; ws += (size_t)BS * E * 2;          // 8 MB  [b,h,d,s]
  _Float16* Qr = (_Float16*)ws; ws += (size_t)BS * E * 2;          // 8 MB  [b,h,s,d]
  _Float16* Kr = (_Float16*)ws; ws += (size_t)BS * E * 2;          // 8 MB  [b,h,s,d]

  const int NX = BS * E, NW = E * E;
  prep_f16<<<(NX + 3 * NW) / 256, 256, 0, stream>>>(x, Wq, Wk, Wv, xh, Wh);

  // 3 proj * 256 row-tiles * 16 col-groups = 12288 waves = 1536 blocks
  qkv_gemm<<<1536, 256, 0, stream>>>(xh, Wh, bq, bk, bv, Qt, Kt, Vt);

  rope_kernel<<<(2 * NX) / 256, 256, 0, stream>>>(Qt, Kt, Qr, Kr);

  attn_kernel<<<Bc * H * (Sq / 128), 256, 0, stream>>>(Qr, Kr, Vt, amsk, out);
}
